// T2ICrossAttentionPool_16630113370426
// MI455X (gfx1250) — matrix operand order repaired and checked by reading, not compiled, not run on hardware
//
#include <hip/hip_runtime.h>

// ---- CDNA5 wave32 WMMA types -------------------------------------------------
typedef __bf16  v16bf __attribute__((ext_vector_type(16)));
typedef __bf16  v4bf  __attribute__((ext_vector_type(4)));
typedef float   v8f   __attribute__((ext_vector_type(8)));

#define WMMA_BF16(a, b, c) \
    __builtin_amdgcn_wmma_f32_16x16x32_bf16(false, (a), false, (b), (short)0, (c), false, false)

// Problem constants (match reference setup_inputs)
#define N_IMG 256
#define N_REG 36
#define N_CAP 256
#define MAX_W 60
#define DIM   1024
#define WP    64        // words padded to 64
#define RP    64        // regions padded to 64
#define DCHUNK 128      // D streamed in 128-wide chunks (8 chunks)
#define LDK   (DCHUNK + 8)   // bf16 row stride: 272B -> bank stride 4, conflict-free frags
#define LDR   (RP + 8)       // bf16 row stride for attn/G: 144B
#define LDS_S (RP + 4)       // f32 row stride for scores/V: 272B

#define SMOOTH 9.0f
#define EPS    1e-8f
#define LEAK   0.1f

__global__ __launch_bounds__(256)
void scan_t2i_pool_kernel(const float* __restrict__ imgs,
                          const float* __restrict__ caps,
                          const int*   __restrict__ cap_lens,
                          float*       __restrict__ out)
{
    // One block per (caption c, image i) pair.
    const int c = blockIdx.x;
    const int i = blockIdx.y;
    const int tid  = threadIdx.x;
    const int wv   = tid >> 5;          // wave id 0..7
    const int lane = tid & 31;
    const int hi   = lane >> 4;         // lane half (0: lanes 0-15, 1: lanes 16-31)
    const int l15  = lane & 15;

    // 4x4 grid of 16x16 output tiles; each wave owns one M-tile row-pair
    const int wt  = wv >> 1;            // M tile (words / gram rows)
    const int rt0 = (wv & 1) * 2;       // N tiles (regions)
    const int rt1 = rt0 + 1;

    __shared__ __bf16 sQ[WP][LDK];      // 17 KB  q chunk     [w][k]  (row-major in K)
    __shared__ __bf16 sP[RP][LDK];      // 17 KB  imgs chunk  [r][k]  (row-major in K)
    __shared__ float  sS[WP][LDS_S];    // 17 KB  raw scores, later V = attn*G
    __shared__ __bf16 sAh[WP][LDR];     //  9 KB  attention (bf16, feeds WMMA)
    __shared__ __bf16 sGh[RP][LDR];     //  9 KB  Gram (bf16, symmetric)
    __shared__ float  sNorm[RP];
    __shared__ float  sW1[WP];          // ||q_w||^2
    __shared__ float  sW12[WP];
    __shared__ float  sW2[WP];          // ||wctx_w||^2
    __shared__ float  sSim[WP];

    const float* capRow = caps + (size_t)c * MAX_W * DIM;
    const float* imgRow = imgs + (size_t)i * N_REG * DIM;
    const int clen = cap_lens[c];

    // ---- phase 0: per-word query norms ||q_w||^2 ----------------------------
    if (tid < WP) sW1[tid] = 0.0f;
    __syncthreads();
    {
        const int w = tid >> 2, sub = tid & 3;   // 4 threads per word row
        float acc = 0.0f;
        if (w < MAX_W && w < clen) {
            const float* row = capRow + w * DIM;
            for (int d = sub * 4; d < DIM; d += 16) {
                float4 v = *reinterpret_cast<const float4*>(row + d);
                acc += v.x * v.x + v.y * v.y + v.z * v.z + v.w * v.w;
            }
        }
        atomicAdd(&sW1[w], acc);
    }

    // ---- phase 1: S = q * P^T  and  G = P * P^T  (K = 1024, bf16 WMMA) ------
    v8f accS0 = {}, accS1 = {}, accG0 = {}, accG1 = {};

    for (int ch = 0; ch < DIM / DCHUNK; ++ch) {
        const int dbase = ch * DCHUNK;
        __syncthreads();
        // stage q chunk: float4 global loads, packed bf16x4 LDS stores
        for (int t = tid; t < WP * (DCHUNK / 4); t += 256) {
            const int w = t >> 5, k = (t & 31) * 4;
            float4 v = make_float4(0.f, 0.f, 0.f, 0.f);
            if (w < MAX_W && w < clen)
                v = *reinterpret_cast<const float4*>(capRow + w * DIM + dbase + k);
            v4bf p;
            p[0] = (__bf16)v.x; p[1] = (__bf16)v.y; p[2] = (__bf16)v.z; p[3] = (__bf16)v.w;
            *reinterpret_cast<v4bf*>(&sQ[w][k]) = p;
        }
        // stage imgs chunk (row-major in K; rows >= 36 zeroed)
        for (int t = tid; t < RP * (DCHUNK / 4); t += 256) {
            const int r = t >> 5, k = (t & 31) * 4;
            float4 v = make_float4(0.f, 0.f, 0.f, 0.f);
            if (r < N_REG)
                v = *reinterpret_cast<const float4*>(imgRow + r * DIM + dbase + k);
            v4bf p;
            p[0] = (__bf16)v.x; p[1] = (__bf16)v.y; p[2] = (__bf16)v.z; p[3] = (__bf16)v.w;
            *reinterpret_cast<v4bf*>(&sP[r][k]) = p;
        }
        __syncthreads();

        #pragma unroll
        for (int kk = 0; kk < DCHUNK / 32; ++kk) {
            const int k0 = kk * 32;
            v16bf a, ag, b0, b1;
            // A fragments: per-lane K runs of 8 are contiguous -> 2x ds_load_b128 each
            #pragma unroll
            for (int e = 0; e < 8; ++e) {
                a[e]     = sQ[wt * 16 + l15][k0 + 8 * hi + e];
                a[e + 8] = sQ[wt * 16 + l15][k0 + 16 + 8 * hi + e];
                ag[e]     = sP[wt * 16 + l15][k0 + 8 * hi + e];
                ag[e + 8] = sP[wt * 16 + l15][k0 + 16 + 8 * hi + e];
            }
            // B fragments: per-lane K runs of 8 (k = 8*hi + e, 16 + 8*hi + e) -> 2x ds_load_b128
            #pragma unroll
            for (int e = 0; e < 8; ++e) {
                b0[e]     = sP[rt0 * 16 + l15][k0 + 8 * hi + e];
                b0[e + 8] = sP[rt0 * 16 + l15][k0 + 16 + 8 * hi + e];
                b1[e]     = sP[rt1 * 16 + l15][k0 + 8 * hi + e];
                b1[e + 8] = sP[rt1 * 16 + l15][k0 + 16 + 8 * hi + e];
            }
            accS0 = WMMA_BF16(a,  b0, accS0);
            accS1 = WMMA_BF16(a,  b1, accS1);
            accG0 = WMMA_BF16(ag, b0, accG0);
            accG1 = WMMA_BF16(ag, b1, accG1);
        }
    }
    __syncthreads();
    // write back S (f32) and G (bf16) — C/D layout: M = v + 8*hi, N = lane&15
    #pragma unroll
    for (int v = 0; v < 8; ++v) {
        const int m = v + 8 * hi;
        sS[wt * 16 + m][rt0 * 16 + l15]  = accS0[v];
        sS[wt * 16 + m][rt1 * 16 + l15]  = accS1[v];
        sGh[wt * 16 + m][rt0 * 16 + l15] = (__bf16)accG0[v];
        sGh[wt * 16 + m][rt1 * 16 + l15] = (__bf16)accG1[v];
    }
    __syncthreads();

    // ---- phase 2: LeakyReLU, per-region L2 norm over words, softmax over r --
    if (tid < RP) {                        // one thread per region column
        float acc = 0.0f;
        #pragma unroll 8
        for (int w = 0; w < WP; ++w) {
            float x = sS[w][tid];
            float l = (x >= 0.0f) ? x : LEAK * x;
            acc += l * l;
        }
        sNorm[tid] = __fsqrt_rn(acc) + EPS;
    }
    __syncthreads();
    if (tid < WP) {                        // one thread per word row
        const int w = tid;
        float vals[N_REG];
        float m = -1e30f;
        #pragma unroll
        for (int r = 0; r < N_REG; ++r) {
            float x = sS[w][r];
            float l = (x >= 0.0f) ? x : LEAK * x;
            float t = SMOOTH * l / sNorm[r];
            vals[r] = t;
            m = fmaxf(m, t);
        }
        float se = 0.0f;
        #pragma unroll
        for (int r = 0; r < N_REG; ++r) { float e = __expf(vals[r] - m); vals[r] = e; se += e; }
        const float inv = 1.0f / se;
        float w12 = 0.0f;
        #pragma unroll
        for (int r = 0; r < N_REG; ++r) {
            float at = vals[r] * inv;
            sAh[w][r] = (__bf16)at;
            w12 += at * sS[w][r];          // w12 = sum_r attn * raw_score
        }
        #pragma unroll
        for (int r = N_REG; r < RP; ++r) sAh[w][r] = (__bf16)0.0f;
        sW12[w] = w12;
    }
    __syncthreads();

    // ---- phase 3: V = attn * G (64x64x64 WMMA); w2^2 = sum_r V .* attn ------
    // G is symmetric, so B(k,n) = G[k][n] = G[n][k] -> row-major contiguous reads.
    v8f accV0 = {}, accV1 = {};
    #pragma unroll
    for (int kk = 0; kk < 2; ++kk) {
        const int k0 = kk * 32;
        v16bf a, b0, b1;
        #pragma unroll
        for (int e = 0; e < 8; ++e) {
            a[e]     = sAh[wt * 16 + l15][k0 + 8 * hi + e];
            a[e + 8] = sAh[wt * 16 + l15][k0 + 16 + 8 * hi + e];
        }
        #pragma unroll
        for (int e = 0; e < 8; ++e) {
            b0[e]     = sGh[rt0 * 16 + l15][k0 + 8 * hi + e];
            b0[e + 8] = sGh[rt0 * 16 + l15][k0 + 16 + 8 * hi + e];
            b1[e]     = sGh[rt1 * 16 + l15][k0 + 8 * hi + e];
            b1[e + 8] = sGh[rt1 * 16 + l15][k0 + 16 + 8 * hi + e];
        }
        accV0 = WMMA_BF16(a, b0, accV0);
        accV1 = WMMA_BF16(a, b1, accV1);
    }
    // sS is dead now; reuse it for V
    #pragma unroll
    for (int v = 0; v < 8; ++v) {
        const int m = v + 8 * hi;
        sS[wt * 16 + m][rt0 * 16 + l15] = accV0[v];
        sS[wt * 16 + m][rt1 * 16 + l15] = accV1[v];
    }
    __syncthreads();
    if (tid < WP) {
        float acc = 0.0f;
        #pragma unroll
        for (int r = 0; r < N_REG; ++r) acc += sS[tid][r] * (float)sAh[tid][r];
        sW2[tid] = acc;                    // ||wctx_w||^2
    }
    __syncthreads();

    // ---- phase 4: cosine sim + masked mean ----------------------------------
    if (tid < WP) {
        const float w1 = __fsqrt_rn(sW1[tid]);
        const float w2 = __fsqrt_rn(fmaxf(sW2[tid], 0.0f));
        sSim[tid] = sW12[tid] / fmaxf(w1 * w2, EPS);
    }
    __syncthreads();
    if (tid == 0) {
        float acc = 0.0f;
        const int n = (clen < MAX_W) ? clen : MAX_W;
        for (int w = 0; w < n; ++w) acc += sSim[w];
        out[(size_t)i * N_CAP + c] = acc / (float)clen;   // out is (I, C) = col.T
    }
}

extern "C" void kernel_launch(void* const* d_in, const int* in_sizes, int n_in,
                              void* d_out, int out_size, void* d_ws, size_t ws_size,
                              hipStream_t stream)
{
    const float* imgs     = (const float*)d_in[0];   // (256, 36, 1024) f32
    const float* caps     = (const float*)d_in[1];   // (256, 60, 1024) f32
    // d_in[2] = img_lens (unused by the reference)
    const int*   cap_lens = (const int*)d_in[3];     // (256,) i32
    float* out = (float*)d_out;                      // (256, 256) f32

    dim3 grid(N_CAP, N_IMG);                         // block = one (caption, image) pair
    scan_t2i_pool_kernel<<<grid, 256, 0, stream>>>(imgs, caps, cap_lens, out);
}
